// MultiHeadSelfAttention_74826920231242
// MI455X (gfx1250) — compile-verified, matrix operand order checked by reading, not run on hardware
//
#include <hip/hip_runtime.h>
#include <hip/hip_bf16.h>

// ---- problem constants (B=2, N=2048, F=1024, H=16, HD=64) ----
#define B_SZ   2
#define N_SEQ  2048
#define F_DIM  1024
#define H_CNT  16
#define HD_DIM 64
#define INV_SQRT_F (1.0f / 32.0f)   // 1/sqrt(1024)

typedef _Float16 v16h __attribute__((ext_vector_type(16)));
typedef _Float16 h8   __attribute__((ext_vector_type(8)));
typedef _Float16 h4   __attribute__((ext_vector_type(4)));
typedef float    v8f  __attribute__((ext_vector_type(8)));

__device__ __forceinline__ v8f wmma_f16(v16h a, v16h b, v8f c) {
  // D = A(16x32 f16) * B(32x16 f16) + C(16x16 f32)
  return __builtin_amdgcn_wmma_f32_16x16x32_f16(false, a, false, b, (short)0, c,
                                                false, false);
}

__device__ __forceinline__ v8f zero8() {
  v8f z;
#pragma unroll
  for (int i = 0; i < 8; ++i) z[i] = 0.0f;
  return z;
}

// A/B fragment load (16-bit, 16x32 over K): per ISA 7.12.2, lanes 0-15 hold
// K = {koff..koff+7, 16+koff..23+koff} with koff = (lane>>4)*8. Two b128 loads.
// For A: rowptr = row m of X. For B: rowptr = row n of W/K/VT (B is loaded
// column-major per lane: lane <-> N, K along VGPRs).
__device__ __forceinline__ v16h frag_ld(const _Float16* rowptr, int k0, int lane) {
  const int koff = (lane >> 4) << 3;
  const _Float16* p = rowptr + k0 + koff;
  h8 lo = *(const h8*)(p);
  h8 hh = *(const h8*)(p + 16);
  v16h a;
#pragma unroll
  for (int i = 0; i < 8; ++i) { a[i] = lo[i]; a[i + 8] = hh[i]; }
  return a;
}

// ---------------- fp32 -> f16 conversion (vectorized x4) ----------------
__global__ __launch_bounds__(256) void cvt_f32_f16(const float* __restrict__ in,
                                                   _Float16* __restrict__ out,
                                                   int n4) {
  int i = blockIdx.x * 256 + threadIdx.x;
  if (i < n4) {
    float4 v = ((const float4*)in)[i];
    h4 o;
    o[0] = (_Float16)v.x; o[1] = (_Float16)v.y;
    o[2] = (_Float16)v.z; o[3] = (_Float16)v.w;
    ((h4*)out)[i] = o;
  }
}

// ---------------- projection GEMM: Y = X @ W^T + b ----------------
// X: [4096, 1024] f16 row-major, W: [1024(out), 1024(in)] f16 row-major.
// Each wave computes a 32x64 tile (2 M-frags x 4 N-frags, 8 f32 accumulators)
// so every B (weight) fragment is reused by two WMMAs and the scheduler has
// 8 independent accumulation chains. mode 0: fp32 row-major to outF.
// mode 1: f16 to [B,H,N,HD]. mode 2: f16 to [B,H,HD,N] (V transposed).
__global__ __launch_bounds__(128) void gemm_proj(const _Float16* __restrict__ X,
                                                 const _Float16* __restrict__ W,
                                                 const float* __restrict__ bias,
                                                 float* __restrict__ outF,
                                                 _Float16* __restrict__ outH,
                                                 int mode) {
  const int lane = threadIdx.x & 31;
  const int wave = threadIdx.x >> 5;
  const int m0 = (blockIdx.y * 4 + wave) * 32;
  const int n0 = blockIdx.x * 64;
  const int r  = lane & 15;
  const int hi = lane >> 4;

  const _Float16* xrow0 = X + (size_t)(m0 + r) * F_DIM;
  const _Float16* xrow1 = X + (size_t)(m0 + 16 + r) * F_DIM;
  const _Float16* wrow  = W + (size_t)(n0 + r) * F_DIM;

  v8f acc[2][4];
#pragma unroll
  for (int mi = 0; mi < 2; ++mi)
#pragma unroll
    for (int t = 0; t < 4; ++t) acc[mi][t] = zero8();

#pragma unroll 2
  for (int k0 = 0; k0 < F_DIM; k0 += 32) {
    // pull the next K-slab toward the WGP$ (global_prefetch_b8; speculative,
    // tail overshoot is silently dropped)
    __builtin_prefetch(xrow0 + k0 + 256, 0, 3);
    __builtin_prefetch(wrow + k0 + 256, 0, 3);
    v16h a0 = frag_ld(xrow0, k0, lane);
    v16h a1 = frag_ld(xrow1, k0, lane);
#pragma unroll
    for (int t = 0; t < 4; ++t) {
      v16h b = frag_ld(wrow + (size_t)t * 16 * F_DIM, k0, lane);
      acc[0][t] = wmma_f16(a0, b, acc[0][t]);
      acc[1][t] = wmma_f16(a1, b, acc[1][t]);
    }
  }

#pragma unroll
  for (int mi = 0; mi < 2; ++mi) {
#pragma unroll
    for (int t = 0; t < 4; ++t) {
      const int col = n0 + t * 16 + r;
      const float bv = bias[col];
#pragma unroll
      for (int rr = 0; rr < 8; ++rr) {
        const int row = m0 + mi * 16 + rr + hi * 8;
        const float v = acc[mi][t][rr] + bv;
        if (mode == 0) {
          outF[(size_t)row * F_DIM + col] = v;
        } else {
          const int bb = row >> 11, ii = row & (N_SEQ - 1);
          const int hh = col >> 6,  dd = col & (HD_DIM - 1);
          size_t dst;
          if (mode == 1)
            dst = (((size_t)(bb * H_CNT + hh)) * N_SEQ + ii) * HD_DIM + dd;
          else
            dst = (((size_t)(bb * H_CNT + hh)) * HD_DIM + dd) * N_SEQ + ii;
          outH[dst] = (_Float16)v;
        }
      }
    }
  }
}

// ---------------- flash attention ----------------
// Q,K: [B,H,N,HD] f16.  VT: [B,H,HD,N] f16.  OH: [B*N, F] f16.
// One wave handles 16 query rows; streams 32 keys per step with online softmax.
__global__ __launch_bounds__(32) void attn_flash(const _Float16* __restrict__ Qg,
                                                 const _Float16* __restrict__ Kg,
                                                 const _Float16* __restrict__ VTg,
                                                 _Float16* __restrict__ OH) {
  __shared__ __align__(16) _Float16 P[16 * 32];

  const int lane = threadIdx.x & 31;
  const int r  = lane & 15;
  const int hi = lane >> 4;
  const int i0 = blockIdx.x * 16;
  const int h  = blockIdx.y;
  const int b  = blockIdx.z;

  const size_t head = (size_t)(b * H_CNT + h);
  const _Float16* Qh  = Qg  + head * N_SEQ * HD_DIM;
  const _Float16* Kh  = Kg  + head * N_SEQ * HD_DIM;
  const _Float16* VTh = VTg + head * HD_DIM * N_SEQ;

  // Q fragments for this 16-row tile (K-dim = HD = 64 -> two 16x32 chunks)
  const _Float16* qrow = Qh + (size_t)(i0 + r) * HD_DIM;
  const v16h qa0 = frag_ld(qrow, 0,  lane);
  const v16h qa1 = frag_ld(qrow, 32, lane);

  float mrun[8], lrun[8];
  v8f o[4];
#pragma unroll
  for (int rr = 0; rr < 8; ++rr) { mrun[rr] = -1e30f; lrun[rr] = 0.0f; }
#pragma unroll
  for (int t = 0; t < 4; ++t) o[t] = zero8();

  for (int j0 = 0; j0 < N_SEQ; j0 += 32) {
    // S = Q * K^T for a 16(M) x 32(J) block: two 16x16 tiles
    const _Float16* krow0 = Kh + (size_t)(j0 + r) * HD_DIM;
    const _Float16* krow1 = Kh + (size_t)(j0 + 16 + r) * HD_DIM;
    v8f s0 = zero8(), s1 = zero8();
    s0 = wmma_f16(qa0, frag_ld(krow0, 0,  lane), s0);
    s0 = wmma_f16(qa1, frag_ld(krow0, 32, lane), s0);
    s1 = wmma_f16(qa0, frag_ld(krow1, 0,  lane), s1);
    s1 = wmma_f16(qa1, frag_ld(krow1, 32, lane), s1);

    // online softmax: row rr lives across the 16 lanes of this half-wave
#pragma unroll
    for (int rr = 0; rr < 8; ++rr) {
      float a0 = s0[rr] * INV_SQRT_F;
      float a1 = s1[rr] * INV_SQRT_F;
      float bm = fmaxf(a0, a1);
#pragma unroll
      for (int msk = 8; msk >= 1; msk >>= 1)
        bm = fmaxf(bm, __shfl_xor(bm, msk, 32));
      const float mn = fmaxf(mrun[rr], bm);
      const float rescale = __expf(mrun[rr] - mn);
      const float p0 = __expf(a0 - mn);
      const float p1 = __expf(a1 - mn);
      float ps = p0 + p1;
#pragma unroll
      for (int msk = 8; msk >= 1; msk >>= 1)
        ps += __shfl_xor(ps, msk, 32);
      lrun[rr] = lrun[rr] * rescale + ps;
      mrun[rr] = mn;
#pragma unroll
      for (int t = 0; t < 4; ++t) o[t][rr] *= rescale;
      // C-layout -> row-major P tile in LDS (for A-fragment reload)
      const int prow = rr + hi * 8;
      P[prow * 32 + r]      = (_Float16)p0;
      P[prow * 32 + 16 + r] = (_Float16)p1;
    }
    __syncthreads();

    // O += P(16x32) * V(32x64): A from LDS, B from transposed V (contig in j)
    const v16h pa = frag_ld(&P[r * 32], 0, lane);
#pragma unroll
    for (int t = 0; t < 4; ++t) {
      const _Float16* vrow = VTh + (size_t)(t * 16 + r) * N_SEQ;
      o[t] = wmma_f16(pa, frag_ld(vrow, j0, lane), o[t]);
    }
    __syncthreads();
  }

  // normalize and store as f16 rows of [B*N, F] for the output projection
#pragma unroll
  for (int t = 0; t < 4; ++t) {
#pragma unroll
    for (int rr = 0; rr < 8; ++rr) {
      const int row = i0 + rr + hi * 8;
      const float v = o[t][rr] / lrun[rr];
      OH[((size_t)(b * N_SEQ + row)) * F_DIM + h * HD_DIM + t * 16 + r] =
          (_Float16)v;
    }
  }
}

// ---------------- launcher ----------------
extern "C" void kernel_launch(void* const* d_in, const int* in_sizes, int n_in,
                              void* d_out, int out_size, void* d_ws, size_t ws_size,
                              hipStream_t stream) {
  (void)in_sizes; (void)n_in; (void)out_size; (void)ws_size;

  const float* x  = (const float*)d_in[0];
  const float* Wq = (const float*)d_in[1];
  const float* bq = (const float*)d_in[2];
  const float* Wk = (const float*)d_in[3];
  const float* bk = (const float*)d_in[4];
  const float* Wv = (const float*)d_in[5];
  const float* bv = (const float*)d_in[6];
  const float* Wo = (const float*)d_in[7];
  const float* bo = (const float*)d_in[8];
  float* out = (float*)d_out;

  char* ws = (char*)d_ws;
  const size_t MB = 1u << 20;
  _Float16* Xh  = (_Float16*)(ws + 0 * MB);   // 8 MB: x f16 [4096,1024]
  _Float16* Wqh = (_Float16*)(ws + 8 * MB);   // 2 MB each
  _Float16* Wkh = (_Float16*)(ws + 10 * MB);
  _Float16* Wvh = (_Float16*)(ws + 12 * MB);
  _Float16* Woh = (_Float16*)(ws + 14 * MB);
  _Float16* Qb  = (_Float16*)(ws + 16 * MB);  // 8 MB  [B,H,N,HD]
  _Float16* Kb  = (_Float16*)(ws + 24 * MB);  // 8 MB  [B,H,N,HD]
  _Float16* VTb = (_Float16*)(ws + 32 * MB);  // 8 MB  [B,H,HD,N]
  _Float16* OHb = (_Float16*)(ws + 40 * MB);  // 8 MB  [B*N, F]

  const int nX = B_SZ * N_SEQ * F_DIM;   // 4 Mi elems
  const int nW = F_DIM * F_DIM;          // 1 Mi elems
  cvt_f32_f16<<<(nX / 4 + 255) / 256, 256, 0, stream>>>(x,  Xh,  nX / 4);
  cvt_f32_f16<<<(nW / 4 + 255) / 256, 256, 0, stream>>>(Wq, Wqh, nW / 4);
  cvt_f32_f16<<<(nW / 4 + 255) / 256, 256, 0, stream>>>(Wk, Wkh, nW / 4);
  cvt_f32_f16<<<(nW / 4 + 255) / 256, 256, 0, stream>>>(Wv, Wvh, nW / 4);
  cvt_f32_f16<<<(nW / 4 + 255) / 256, 256, 0, stream>>>(Wo, Woh, nW / 4);

  // 32x64 tile per wave, 4 waves per block -> 128 M-rows per block
  dim3 gg(F_DIM / 64, (B_SZ * N_SEQ) / 128);  // (16, 32)
  gemm_proj<<<gg, 128, 0, stream>>>(Xh, Wqh, bq, nullptr, Qb,  1);
  gemm_proj<<<gg, 128, 0, stream>>>(Xh, Wkh, bk, nullptr, Kb,  1);
  gemm_proj<<<gg, 128, 0, stream>>>(Xh, Wvh, bv, nullptr, VTb, 2);

  attn_flash<<<dim3(N_SEQ / 16, H_CNT, B_SZ), 32, 0, stream>>>(Qb, Kb, VTb, OHb);

  gemm_proj<<<gg, 128, 0, stream>>>(OHb, Woh, bo, out, nullptr, 0);
}
